// MySelfAttention_18391049961927
// MI455X (gfx1250) — matrix operand order repaired and checked by reading, not compiled, not run on hardware
//
#include <hip/hip_runtime.h>

// Problem constants (from reference): B=8, S=2048, D=OUT=768
#define BATCH 8
#define S_LEN 2048
#define DIM   768

// Flip to 0 if the async-to-LDS inline asm fails to assemble.
#define USE_ASYNC_Q 1

typedef __attribute__((ext_vector_type(16))) __bf16 v16bf;
typedef __attribute__((ext_vector_type(8)))  __bf16 v8bf;
typedef __attribute__((ext_vector_type(8)))  float  v8f;

// Build a 16-element bf16 fragment from two 16-byte chunks.
// Matches CDNA5 16-bit A/B layouts where each lane holds two 8-element
// contiguous K runs.
static __device__ __forceinline__ v16bf load_frag16(const __bf16* p0, const __bf16* p1) {
    v8bf lo = *(const v8bf*)p0;
    v8bf hi = *(const v8bf*)p1;
    v16bf r;
#pragma unroll
    for (int i = 0; i < 8; ++i) { r[i] = lo[i]; r[i + 8] = hi[i]; }
    return r;
}

// ---------------------------------------------------------------------------
// Kernel 1: QKV projection GEMM.  C[M=16384, N=768] = X[M,768] * W[z][768,768]
// Block tile 128x128, K-chunk 32, bf16 WMMA 16x16x32, f32 accumulate.
// z=0 -> Q (scaled by 1/sqrt(768)), z=1 -> K, z=2 -> V stored transposed.
// ---------------------------------------------------------------------------
__global__ __launch_bounds__(256) void qkv_gemm(const float* __restrict__ X,
                                                const float* __restrict__ W,
                                                __bf16* __restrict__ Qb,
                                                __bf16* __restrict__ Kb,
                                                __bf16* __restrict__ VT,
                                                float qscale)
{
    constexpr int LDA = 40;                       // padded K-stride (bank-conflict pad)
    __shared__ alignas(16) __bf16 As[128 * LDA];  // A tile: [row][k]
    __shared__ alignas(16) __bf16 Bs[128 * LDA];  // B tile transposed: [n][k]

    const int tid  = threadIdx.x;
    const int lane = tid & 31;
    const int wave = tid >> 5;
    const int nbase  = blockIdx.x * 128;
    const size_t mbase = (size_t)blockIdx.y * 128;
    const int wsel   = blockIdx.z;
    const float* Wm  = W + (size_t)wsel * DIM * DIM;

    v8f acc[8];
#pragma unroll
    for (int i = 0; i < 8; ++i)
#pragma unroll
        for (int j = 0; j < 8; ++j) acc[i][j] = 0.0f;

    const int r16 = lane & 15;
    const int hi  = lane >> 4;

    for (int kk = 0; kk < DIM; kk += 32) {
        __syncthreads();
        // ---- stage A tile 128x32 (fp32 -> bf16) ----
        {
            const int row = tid >> 1;
            const int k0  = (tid & 1) * 16;
            const float* src = X + (mbase + row) * (size_t)DIM + kk + k0;
            __bf16* dst = &As[row * LDA + k0];
#pragma unroll
            for (int v = 0; v < 4; ++v) {
                float4 f = ((const float4*)src)[v];
                dst[v * 4 + 0] = (__bf16)f.x;
                dst[v * 4 + 1] = (__bf16)f.y;
                dst[v * 4 + 2] = (__bf16)f.z;
                dst[v * 4 + 3] = (__bf16)f.w;
            }
        }
        // ---- stage B tile 32x128 transposed -> Bs[n][k] ----
        {
            const int k  = tid >> 3;
            const int n0 = (tid & 7) * 16;
            const float* src = Wm + (size_t)(kk + k) * DIM + nbase + n0;
#pragma unroll
            for (int v = 0; v < 4; ++v) {
                float4 f = ((const float4*)src)[v];
                Bs[(n0 + v * 4 + 0) * LDA + k] = (__bf16)f.x;
                Bs[(n0 + v * 4 + 1) * LDA + k] = (__bf16)f.y;
                Bs[(n0 + v * 4 + 2) * LDA + k] = (__bf16)f.z;
                Bs[(n0 + v * 4 + 3) * LDA + k] = (__bf16)f.w;
            }
        }
        __syncthreads();

        // ---- compute: wave owns rows [wave*16, wave*16+16), all 128 cols ----
        const int arow = wave * 16 + r16;
        const int ak   = hi * 8;
        v16bf afrag = load_frag16(&As[arow * LDA + ak], &As[arow * LDA + ak + 16]);
#pragma unroll
        for (int nt = 0; nt < 8; ++nt) {
            const int bn = nt * 16 + r16;
            const int bk = hi * 8;
            v16bf bfrag = load_frag16(&Bs[bn * LDA + bk], &Bs[bn * LDA + bk + 16]);
            acc[nt] = __builtin_amdgcn_wmma_f32_16x16x32_bf16(
                false, afrag, false, bfrag, (short)0, acc[nt], false, false);
        }
    }

    // ---- writeback (bf16).  C layout: VGPR i -> row i + (lane>=16 ? 8 : 0) ----
    const int rlo = hi * 8;
#pragma unroll
    for (int nt = 0; nt < 8; ++nt) {
#pragma unroll
        for (int i = 0; i < 8; ++i) {
            const size_t grow = mbase + wave * 16 + rlo + i;   // flat b*S+s
            const int    gcol = nbase + nt * 16 + r16;
            const float  v    = acc[nt][i];
            if (wsel == 0) {
                Qb[grow * DIM + gcol] = (__bf16)(v * qscale);
            } else if (wsel == 1) {
                Kb[grow * DIM + gcol] = (__bf16)v;
            } else {
                const size_t bb = grow >> 11;     // / 2048
                const size_t ss = grow & 2047;
                VT[(bb * DIM + gcol) * (size_t)S_LEN + ss] = (__bf16)v;
            }
        }
    }
}

// ---------------------------------------------------------------------------
// Kernel 2: flash attention.  Block = (32 queries, one batch), 8 waves.
// Wave (rg = w>>2, cg = w&3):
//   scores: S-tile rows [rg*16,+16) x keys [cg*16,+16), full 768 K-dim.
//   P@V   : rows [rg*16,+16) x out-cols [cg*192,+192) (12 C-tiles).
// Online softmax over 64-key chunks.
// ---------------------------------------------------------------------------
__global__ __launch_bounds__(256) void flash_attn(const __bf16* __restrict__ Qb,
                                                  const __bf16* __restrict__ Kb,
                                                  const __bf16* __restrict__ VT,
                                                  float* __restrict__ Out)
{
    __shared__ alignas(16) __bf16 Qs[32 * DIM];   // 48 KB: query tile, bf16
    __shared__ alignas(16) float  Ss[32 * 64];    // 8 KB: score tile
    __shared__ alignas(16) __bf16 Ps[32 * 64];    // 4 KB: probabilities (A-frag source)
    __shared__ float m_s[32], l_s[32], sc_s[32];  // running max / sum / rescale

    const int tid  = threadIdx.x;
    const int lane = tid & 31;
    const int wave = tid >> 5;
    const int rg   = wave >> 2;     // row group: 0 or 1
    const int cg   = wave & 3;      // column group: 0..3
    const int b    = blockIdx.y;
    const int q0   = blockIdx.x * 32;

    // ---- stage Q tile (bf16, pre-scaled): 32*768*2 = 48KB = 3072 x 16B ----
#if USE_ASYNC_Q
    {
        const __bf16* src = Qb + ((size_t)b * S_LEN + q0) * DIM;
        const unsigned lbase = (unsigned)(unsigned long long)&Qs[0];
#pragma unroll
        for (int i = 0; i < 12; ++i) {
            const unsigned          laddr = lbase + tid * 16 + i * 4096;
            const unsigned long long gaddr = (unsigned long long)src + tid * 16 + i * 4096;
            asm volatile("global_load_async_to_lds_b128 %0, %1, off"
                         :: "v"(laddr), "v"(gaddr) : "memory");
        }
#if __has_builtin(__builtin_amdgcn_s_wait_asynccnt)
        __builtin_amdgcn_s_wait_asynccnt(0);
#else
        asm volatile("s_wait_asynccnt 0x0" ::: "memory");
#endif
    }
#else
    {
        const uint4* src = (const uint4*)(Qb + ((size_t)b * S_LEN + q0) * DIM);
        uint4* dst = (uint4*)Qs;
#pragma unroll
        for (int i = 0; i < 12; ++i) dst[tid + i * 256] = src[tid + i * 256];
    }
#endif
    if (tid < 32) { m_s[tid] = -1e30f; l_s[tid] = 0.0f; }

    v8f oacc[12];
#pragma unroll
    for (int ct = 0; ct < 12; ++ct)
#pragma unroll
        for (int j = 0; j < 8; ++j) oacc[ct][j] = 0.0f;

    const int r16 = lane & 15;
    const int hi  = lane >> 4;

    for (int kc = 0; kc < S_LEN; kc += 64) {
        __syncthreads();

        // ---- scores: 24 WMMAs per wave, K fragments straight from global ----
        {
            v8f sacc;
#pragma unroll
            for (int j = 0; j < 8; ++j) sacc[j] = 0.0f;
            const __bf16* kbase = Kb + ((size_t)b * S_LEN + kc + cg * 16 + r16) * DIM;
            if (kc + 64 < S_LEN) __builtin_prefetch(kbase + (size_t)64 * DIM, 0, 0);
            const __bf16* qrow = &Qs[(rg * 16 + r16) * DIM];
#pragma unroll
            for (int ks = 0; ks < 24; ++ks) {
                const int k0 = ks * 32;
                v16bf af  = load_frag16(qrow + k0 + hi * 8, qrow + k0 + 16 + hi * 8);
                v16bf bf_ = load_frag16(kbase + k0 + hi * 8, kbase + k0 + 16 + hi * 8);
                sacc = __builtin_amdgcn_wmma_f32_16x16x32_bf16(
                    false, af, false, bf_, (short)0, sacc, false, false);
            }
#pragma unroll
            for (int i = 0; i < 8; ++i)
                Ss[(rg * 16 + i + hi * 8) * 64 + cg * 16 + r16] = sacc[i];
        }
        __syncthreads();

        // ---- online softmax: 32 rows handled by the 32 lanes of wave 0 ----
        if (tid < 32) {
            const int row = tid;
            const float mold = m_s[row];
            float mx = mold;
#pragma unroll 8
            for (int j = 0; j < 64; ++j) mx = fmaxf(mx, Ss[row * 64 + j]);
            const float sc = __expf(mold - mx);
            float l = l_s[row] * sc;
#pragma unroll 8
            for (int j = 0; j < 64; ++j) {
                const float p = __expf(Ss[row * 64 + j] - mx);
                l += p;
                Ps[row * 64 + j] = (__bf16)p;
            }
            m_s[row] = mx; l_s[row] = l; sc_s[row] = sc;
        }
        __syncthreads();

        // ---- rescale accumulators, then P @ V (12 tiles, 24 WMMAs) ----
        {
#pragma unroll
            for (int i = 0; i < 8; ++i) {
                const float sc = sc_s[rg * 16 + i + hi * 8];
#pragma unroll
                for (int ct = 0; ct < 12; ++ct) oacc[ct][i] *= sc;
            }
            const int colb = cg * 192;
            const __bf16* prow = &Ps[(rg * 16 + r16) * 64];
#pragma unroll
            for (int ks2 = 0; ks2 < 2; ++ks2) {
                v16bf af = load_frag16(prow + ks2 * 32 + hi * 8,
                                       prow + ks2 * 32 + 16 + hi * 8);
#pragma unroll
                for (int ct = 0; ct < 12; ++ct) {
                    const __bf16* vb = VT + ((size_t)b * DIM + colb + ct * 16 + r16) * (size_t)S_LEN
                                          + kc + ks2 * 32 + hi * 8;
                    v16bf bf_ = load_frag16(vb, vb + 16);
                    oacc[ct] = __builtin_amdgcn_wmma_f32_16x16x32_bf16(
                        false, af, false, bf_, (short)0, oacc[ct], false, false);
                }
            }
        }
    }

    // ---- epilogue: normalize by running sum, store fp32 ----
    __syncthreads();
#pragma unroll
    for (int i = 0; i < 8; ++i) {
        const float linv = 1.0f / l_s[rg * 16 + i + hi * 8];
#pragma unroll
        for (int ct = 0; ct < 12; ++ct) {
            const size_t grow = (size_t)b * S_LEN + q0 + rg * 16 + i + hi * 8;
            const int    col  = cg * 192 + ct * 16 + r16;
            Out[grow * DIM + col] = oacc[ct][i] * linv;
        }
    }
}

// ---------------------------------------------------------------------------
extern "C" void kernel_launch(void* const* d_in, const int* in_sizes, int n_in,
                              void* d_out, int out_size, void* d_ws, size_t ws_size,
                              hipStream_t stream) {
    (void)in_sizes; (void)n_in; (void)out_size; (void)ws_size;
    const float* x = (const float*)d_in[0];   // [B,S,D] fp32
    const float* W = (const float*)d_in[1];   // [3,D,OUT] fp32
    float* out = (float*)d_out;               // [B,S,OUT] fp32

    const size_t per = (size_t)BATCH * S_LEN * DIM;
    __bf16* qb = (__bf16*)d_ws;               // [B,S,OUT] bf16 (pre-scaled)
    __bf16* kb = qb + per;                    // [B,S,OUT] bf16
    __bf16* vt = kb + per;                    // [B,OUT,S] bf16 (transposed V)

    const float qscale = 0.03608439182435161f;  // 1/sqrt(768)

    dim3 g1(DIM / 128, (BATCH * S_LEN) / 128, 3);
    qkv_gemm<<<g1, 256, 0, stream>>>(x, W, qb, kb, vt, qscale);

    dim3 g2(S_LEN / 32, BATCH);
    flash_attn<<<g2, 256, 0, stream>>>(qb, kb, vt, out);
}
